// Transducer_53085795778976
// MI455X (gfx1250) — matrix operand order repaired and checked by reading, not compiled, not run on hardware
//
#include <hip/hip_runtime.h>

// ---------------- problem constants ----------------
#define B_   4
#define T_   256
#define U_   64
#define U1_  65          // U+1
#define DJ   512         // D_ENC = D_DEC = D_J
#define V_   1024
#define KC   16          // DJ / 32 k-chunks per WMMA K loop
#define VT   64          // V_ / 16 v tiles
#define UT   5           // ceil(65/16) u tiles

typedef __bf16 bf16;
typedef __attribute__((ext_vector_type(8)))  bf16  v8bf;
typedef __attribute__((ext_vector_type(16))) bf16  v16bf;
typedef __attribute__((ext_vector_type(8)))  float v8f;

// CDNA5 hardware tanh (trans pipe).  v_nops cover the TRANS-op result hazard.
static __device__ __forceinline__ float htanh(float x) {
  float r;
  asm volatile("v_tanh_f32 %0, %1\n\tv_nop\n\tv_nop" : "=v"(r) : "v"(x));
  return r;
}

// ---------------------------------------------------------------------------
// A-fragment loader: 16-bit A matrix 16x32 (M x K) from an LDS tile laid out
// [16 rows][DJ bf16].  Contiguous k order (not the ISA table order):
//   lanes 0-15 : row M = lane,    K = 0..15
//   lanes 16-31: row M = lane-16, K = 16..31
// ---------------------------------------------------------------------------
static __device__ __forceinline__ v16bf load_afrag(const bf16* tile, int lane, int kc) {
  const bf16* p = tile + (lane & 15) * DJ + kc * 32 + (lane >> 4) * 16;
  v8bf lo = *(const v8bf*)(p);
  v8bf hi = *(const v8bf*)(p + 8);
  return __builtin_shufflevector(lo, hi, 0,1,2,3,4,5,6,7,8,9,10,11,12,13,14,15);
}

// ---------------------------------------------------------------------------
// Weight pre-pack: W is [N, K=DJ] fp32 row-major.  Packed B-fragment layout (contiguous k, as load_afrag) for
// 16-bit B (K=32 x N=16): lane n<16: column n, K=0..15; lane n>=16: column
// n-16, K=16..31.  Each lane's 16 bf16 (32B) contiguous.
// ---------------------------------------------------------------------------
__global__ void pack_w_kernel(const float* __restrict__ W, bf16* __restrict__ pack, int ntiles) {
  int idx = blockIdx.x * blockDim.x + threadIdx.x;
  int total = ntiles * KC * 32;
  if (idx >= total) return;
  int lane = idx & 31;
  int frag = idx >> 5;
  int kc   = frag % KC;
  int nt   = frag / KC;
  int col  = nt * 16 + (lane & 15);
  int k0   = kc * 32 + (lane >> 4) * 16;
  const float* src = W + (size_t)col * DJ + k0;
  bf16* dst = pack + ((size_t)idx << 4);
#pragma unroll
  for (int i = 0; i < 16; ++i) dst[i] = (bf16)src[i];
}

// ---------------------------------------------------------------------------
// 16-row projection GEMM + bias: C[m,n] = sum_k A[m,k]*W[n,k] + bias[n]
// Double-buffered A (LDS) and B (global) operand streams.
// ---------------------------------------------------------------------------
__global__ __launch_bounds__(256) void gemm_bias_kernel(
    const float* __restrict__ A, int M,
    const bf16* __restrict__ wpack,
    const float* __restrict__ bias,
    float* __restrict__ C) {
  __shared__ __align__(16) bf16 aT[16 * DJ];
  const int N = DJ;
  int tid = threadIdx.x;
  int m0  = blockIdx.x * 16;

  for (int idx = tid; idx < 16 * DJ; idx += 256) {
    int r = idx >> 9;
    int k = idx & (DJ - 1);
    int row = m0 + r; if (row >= M) row = M - 1;
    aT[r * DJ + k] = (bf16)A[(size_t)row * DJ + k];
  }
  __syncthreads();

  int lane = tid & 31, wave = tid >> 5, half = lane >> 4;

  v8f acc[4];
  v8f zero = {0.f,0.f,0.f,0.f,0.f,0.f,0.f,0.f};
#pragma unroll
  for (int i = 0; i < 4; ++i) acc[i] = zero;

  const char* bp = (const char*)wpack + ((size_t)(wave * 4) * KC * 32 + lane) * 32;

  v16bf acur = load_afrag(aT, lane, 0);
  v16bf bcur = *(const v16bf*)bp;

  for (int kc = 0; kc < KC; ++kc) {
    int kn = (kc + 1 < KC) ? kc + 1 : kc;
    v16bf anxt = load_afrag(aT, lane, kn);      // issued before this kc's WMMAs
    const char* bpn = bp + 1024;                // next kc fragment base
#pragma unroll
    for (int i = 0; i < 4; ++i) {
      v16bf bnxt = (i < 3) ? *(const v16bf*)(bp + (size_t)(i + 1) * (KC * 1024))
                           : *(const v16bf*)(bpn);   // wrap: stays in-bounds, unused at end
      acc[i] = __builtin_amdgcn_wmma_f32_16x16x32_bf16(false, acur, false, bcur, (short)0, acc[i], false, false);
      bcur = bnxt;
    }
    acur = anxt;
    bp = bpn;
  }

#pragma unroll
  for (int i = 0; i < 4; ++i) {
    int col = (wave * 4 + i) * 16 + (lane & 15);
    float bv = bias[col];
#pragma unroll
    for (int r = 0; r < 8; ++r) {
      int row = m0 + half * 8 + r;           // C layout: VGPR r -> M = r + 8*half
      if (row < M) C[(size_t)row * N + col] = acc[i][r] + bv;
    }
  }
}

// ---------------------------------------------------------------------------
// Fused joint + log-softmax-select kernel.
// One WG per (b, t-pair, u-tile of 16): two 16x512 h tiles in LDS, 512
// threads = 16 waves x 4 v-tiles.  Every B fragment feeds TWO WMMAs; both
// operand streams double-buffered so waits carry a full iteration of slack.
// ---------------------------------------------------------------------------
__global__ __launch_bounds__(512) void joint_softmax_kernel(
    const float* __restrict__ enc_h,        // [B*T, DJ]
    const float* __restrict__ dec_h,        // [B*U1, DJ]
    const bf16*  __restrict__ wproj_pack,   // [VT*KC*32*16]
    const int*   __restrict__ targets,      // [B, U]
    float* __restrict__ blank_out,          // [B*T, U1]
    float* __restrict__ emit_out) {         // [B*T, U]
  __shared__ __align__(16) bf16 hT[2][16 * DJ];
  __shared__ float red_m[2][16][16];
  __shared__ float red_s[2][16][16];
  __shared__ float cap_b[2][16];
  __shared__ float cap_e[2][16];

  int tid = threadIdx.x;
  int blk = blockIdx.x;
  int ut  = blk % UT;
  int tp  = (blk / UT) % (T_ / 2);
  int b   = blk / (UT * (T_ / 2));
  int u0  = ut * 16;
  int t0  = tp * 2;

  if (tid < 32) { cap_b[tid >> 4][tid & 15] = 0.f; cap_e[tid >> 4][tid & 15] = 0.f; }

  const float* eh0 = enc_h + (size_t)(b * T_ + t0) * DJ;
  const float* dh  = dec_h + (size_t)b * U1_ * DJ;

  // stage two h tiles: h[rt][r][j] = tanh(enc_h[t0+rt][j] * dec_h[u0+r][j])
  for (int idx = tid; idx < 2 * 16 * DJ; idx += 512) {
    int rt = idx >> 13;               // /8192
    int r  = (idx >> 9) & 15;
    int j  = idx & (DJ - 1);
    int u = u0 + r; if (u > U_) u = U_;     // clamp pad rows (stores masked later)
    hT[rt][r * DJ + j] = (bf16)htanh(eh0[(size_t)rt * DJ + j] * dh[(size_t)u * DJ + j]);
  }
  __syncthreads();

  int lane = tid & 31, wave = tid >> 5, half = lane >> 4;

  int tgt[8];
#pragma unroll
  for (int r = 0; r < 8; ++r) {
    int u = u0 + half * 8 + r;
    tgt[r] = (u < U_) ? targets[b * U_ + u] : -1;
  }

  v8f acc0[4], acc1[4];
  v8f zero = {0.f,0.f,0.f,0.f,0.f,0.f,0.f,0.f};
#pragma unroll
  for (int i = 0; i < 4; ++i) { acc0[i] = zero; acc1[i] = zero; }

  // byte address of this lane's fragment for (vt = wave*4, kc = 0);
  // fragment (i, kc) lives at + (i*KC + kc) * 1024 bytes (immediate offsets).
  const char* bp = (const char*)wproj_pack + ((size_t)(wave * 4) * KC * 32 + lane) * 32;

  v16bf a0 = load_afrag(hT[0], lane, 0);
  v16bf a1 = load_afrag(hT[1], lane, 0);
  v16bf bcur = *(const v16bf*)bp;

  for (int kc = 0; kc < KC; ++kc) {
    int kn = (kc + 1 < KC) ? kc + 1 : kc;
    v16bf a0n = load_afrag(hT[0], lane, kn);    // ds loads issued ahead of WMMAs
    v16bf a1n = load_afrag(hT[1], lane, kn);
    const char* bpn = bp + 1024;                // next kc fragment base
#pragma unroll
    for (int i = 0; i < 4; ++i) {
      v16bf bnxt = (i < 3) ? *(const v16bf*)(bp + (size_t)(i + 1) * (KC * 1024))
                           : *(const v16bf*)(bpn);   // wrap: in-bounds, unused after last kc
      acc0[i] = __builtin_amdgcn_wmma_f32_16x16x32_bf16(false, a0, false, bcur, (short)0, acc0[i], false, false);
      acc1[i] = __builtin_amdgcn_wmma_f32_16x16x32_bf16(false, a1, false, bcur, (short)0, acc1[i], false, false);
      bcur = bnxt;
    }
    a0 = a0n; a1 = a1n;
    bp = bpn;
  }

  // two-pass softmax over the 4 (v-tiles) x 8 (rows) register tile, per rt
  float m0[8], s0[8], m1[8], s1[8];
#pragma unroll
  for (int r = 0; r < 8; ++r) {
    m0[r] = -__builtin_inff(); s0[r] = 0.f;
    m1[r] = -__builtin_inff(); s1[r] = 0.f;
  }
#pragma unroll
  for (int i = 0; i < 4; ++i)
#pragma unroll
    for (int r = 0; r < 8; ++r) {
      m0[r] = fmaxf(m0[r], acc0[i][r]);
      m1[r] = fmaxf(m1[r], acc1[i][r]);
    }
#pragma unroll
  for (int i = 0; i < 4; ++i) {
    int col = (wave * 4 + i) * 16 + (lane & 15);
#pragma unroll
    for (int r = 0; r < 8; ++r) {
      float v0 = acc0[i][r], v1 = acc1[i][r];
      s0[r] += __expf(v0 - m0[r]);
      s1[r] += __expf(v1 - m1[r]);
      int row = half * 8 + r;
      if (col == 0)      { cap_b[0][row] = v0; cap_b[1][row] = v1; }
      if (col == tgt[r]) { cap_e[0][row] = v0; cap_e[1][row] = v1; }
    }
  }

  // butterfly (m,s) reduce across the 16 lanes that share each row
  for (int mask = 1; mask < 16; mask <<= 1) {
#pragma unroll
    for (int r = 0; r < 8; ++r) {
      float mo = __shfl_xor(m0[r], mask, 32);
      float so = __shfl_xor(s0[r], mask, 32);
      float mn = fmaxf(m0[r], mo);
      s0[r] = s0[r] * __expf(m0[r] - mn) + so * __expf(mo - mn);
      m0[r] = mn;
      mo = __shfl_xor(m1[r], mask, 32);
      so = __shfl_xor(s1[r], mask, 32);
      mn = fmaxf(m1[r], mo);
      s1[r] = s1[r] * __expf(m1[r] - mn) + so * __expf(mo - mn);
      m1[r] = mn;
    }
  }
  if ((lane & 15) == 0) {
#pragma unroll
    for (int r = 0; r < 8; ++r) {
      int row = half * 8 + r;
      red_m[0][wave][row] = m0[r];  red_s[0][wave][row] = s0[r];
      red_m[1][wave][row] = m1[r];  red_s[1][wave][row] = s1[r];
    }
  }
  __syncthreads();

  if (tid < 32) {
    int rt = tid >> 4, rr = tid & 15;
    float M = -__builtin_inff();
#pragma unroll
    for (int w = 0; w < 16; ++w) M = fmaxf(M, red_m[rt][w][rr]);
    float S = 0.f;
#pragma unroll
    for (int w = 0; w < 16; ++w) S += red_s[rt][w][rr] * __expf(red_m[rt][w][rr] - M);
    float denom = M + __logf(S);
    int u = u0 + rr;
    int t = t0 + rt;
    if (u < U1_) blank_out[(size_t)(b * T_ + t) * U1_ + u] = cap_b[rt][rr] - denom;
    if (u < U_)  emit_out [(size_t)(b * T_ + t) * U_  + u] = cap_e[rt][rr] - denom;
  }
}

// ---------------------------------------------------------------------------
// Alpha forward DP: single wave per batch, zero barriers.
// Lane l owns u = 3l .. 3l+2 of the 65-wide lattice column axis.
// Scans = serial in-chunk + __shfl_up wave scan (inclusive -> shifted excl).
// ---------------------------------------------------------------------------
static __device__ __forceinline__ float lsef(float a, float b) {
  float mn = fmaxf(a, b);
  return (mn == -__builtin_inff()) ? mn
       : mn + __logf(__expf(a - mn) + __expf(b - mn));
}

static __device__ __forceinline__ float wscan_add(float x, int l) {
#pragma unroll
  for (int off = 1; off < 32; off <<= 1) {
    float v = __shfl_up(x, off, 32);
    if (l >= off) x += v;
  }
  return x;
}

static __device__ __forceinline__ float wscan_lse(float x, int l) {
#pragma unroll
  for (int off = 1; off < 32; off <<= 1) {
    float v = __shfl_up(x, off, 32);
    if (l >= off) x = lsef(x, v);
  }
  return x;
}

__global__ __launch_bounds__(32) void rnnt_dp_kernel(
    const float* __restrict__ blank,   // [B*T, U1]
    const float* __restrict__ emit,    // [B*T, U]
    const int*   __restrict__ inputs_length,
    const int*   __restrict__ targets_length,
    float* __restrict__ finals) {      // [B]
  int b = blockIdx.x;
  int l = threadIdx.x;
  const float* Bp = blank + (size_t)b * T_ * U1_;
  const float* Ep = emit  + (size_t)b * T_ * U_;
  int t_idx = inputs_length[b] - 1;
  int u_idx = targets_length[b];
  int u0c = l * 3;
  const float NINF = -__builtin_inff();

  float alpha[3];
  // t = 0: alpha[u] = sum_{j<u} emit[0,j]  (inclusive scan of shifted emit)
  {
    float x[3];
#pragma unroll
    for (int i = 0; i < 3; ++i) {
      int u = u0c + i;
      x[i] = (u >= 1 && u < U1_) ? Ep[u - 1] : 0.f;
    }
    float c0 = x[0], c1 = c0 + x[1], c2 = c1 + x[2];
    float incl = wscan_add(c2, l);
    float excl = __shfl_up(incl, 1, 32); if (l == 0) excl = 0.f;
    alpha[0] = c0 + excl; alpha[1] = c1 + excl; alpha[2] = c2 + excl;
  }
  if (t_idx == 0) {
#pragma unroll
    for (int i = 0; i < 3; ++i) {
      int u = u0c + i;
      if (u == u_idx) finals[b] = alpha[i] + Bp[u];
    }
  }

  for (int t = 1; t < T_; ++t) {
    float x[3], tmpv[3];
#pragma unroll
    for (int i = 0; i < 3; ++i) {
      int u = u0c + i;
      bool val = (u < U1_);
      tmpv[i] = alpha[i] + (val ? Bp[(size_t)(t - 1) * U1_ + u] : 0.f);
      x[i] = (u >= 1 && val) ? Ep[(size_t)t * U_ + (u - 1)] : 0.f;
    }
    // cem = inclusive add-scan of x
    float c0 = x[0], c1 = c0 + x[1], c2 = c1 + x[2];
    float inca = wscan_add(c2, l);
    float exca = __shfl_up(inca, 1, 32); if (l == 0) exca = 0.f;
    float cem0 = c0 + exca, cem1 = c1 + exca, cem2 = c2 + exca;
    // y = tmp - cem (identity -inf on invalid slots)
    float y0 = (u0c + 0 < U1_) ? tmpv[0] - cem0 : NINF;
    float y1 = (u0c + 1 < U1_) ? tmpv[1] - cem1 : NINF;
    float y2 = (u0c + 2 < U1_) ? tmpv[2] - cem2 : NINF;
    // z = inclusive LSE-scan of y
    float z0 = y0, z1 = lsef(z0, y1), z2 = lsef(z1, y2);
    float incl = wscan_lse(z2, l);
    float excl = __shfl_up(incl, 1, 32); if (l == 0) excl = NINF;
    alpha[0] = lsef(z0, excl) + cem0;
    alpha[1] = lsef(z1, excl) + cem1;
    alpha[2] = lsef(z2, excl) + cem2;
    if (t == t_idx) {
#pragma unroll
      for (int i = 0; i < 3; ++i) {
        int u = u0c + i;
        if (u == u_idx) finals[b] = alpha[i] + Bp[(size_t)t * U1_ + u];
      }
    }
  }
}

__global__ void finalize_kernel(const float* __restrict__ finals, float* __restrict__ out) {
  out[0] = -0.25f * (finals[0] + finals[1] + finals[2] + finals[3]);
}

// ---------------------------------------------------------------------------
extern "C" void kernel_launch(void* const* d_in, const int* in_sizes, int n_in,
                              void* d_out, int out_size, void* d_ws, size_t ws_size,
                              hipStream_t stream) {
  (void)in_sizes; (void)n_in; (void)out_size; (void)ws_size;
  const float* enc_state      = (const float*)d_in[0];  // [B,T,512]
  const float* dec_state      = (const float*)d_in[1];  // [B,65,512]
  const int*   targets        = (const int*)  d_in[2];  // [B,64]
  const int*   inputs_length  = (const int*)  d_in[3];  // [B]
  const int*   targets_length = (const int*)  d_in[4];  // [B]
  const float* W_enc          = (const float*)d_in[5];  // [512,512]
  const float* b_enc          = (const float*)d_in[6];  // [512]
  const float* W_dec          = (const float*)d_in[7];  // [512,512]
  const float* b_dec          = (const float*)d_in[8];  // [512]
  const float* W_proj         = (const float*)d_in[9];  // [1024,512]

  char* ws = (char*)d_ws;
  size_t off = 0;
  auto alloc = [&](size_t bytes) -> void* {
    void* p = ws + off;
    off = (off + bytes + 255) & ~(size_t)255;
    return p;
  };
  float* enc_h  = (float*)alloc((size_t)B_ * T_ * DJ * sizeof(float));
  float* dec_h  = (float*)alloc((size_t)B_ * U1_ * DJ * sizeof(float));
  bf16*  wencp  = (bf16*) alloc((size_t)DJ * DJ * 2);
  bf16*  wdecp  = (bf16*) alloc((size_t)DJ * DJ * 2);
  bf16*  wprojp = (bf16*) alloc((size_t)V_ * DJ * 2);
  float* blankL = (float*)alloc((size_t)B_ * T_ * U1_ * sizeof(float));
  float* emitL  = (float*)alloc((size_t)B_ * T_ * U_ * sizeof(float));
  float* finals = (float*)alloc((size_t)B_ * sizeof(float));

  // 1) pre-pack weights into bf16 B-fragment layout
  pack_w_kernel<<<(32 * KC * 32 + 255) / 256, 256, 0, stream>>>(W_enc, wencp, 32);
  pack_w_kernel<<<(32 * KC * 32 + 255) / 256, 256, 0, stream>>>(W_dec, wdecp, 32);
  pack_w_kernel<<<(VT * KC * 32 + 255) / 256, 256, 0, stream>>>(W_proj, wprojp, VT);

  // 2) enc_h / dec_h projections (WMMA)
  gemm_bias_kernel<<<(B_ * T_) / 16, 256, 0, stream>>>(enc_state, B_ * T_, wencp, b_enc, enc_h);
  gemm_bias_kernel<<<(B_ * U1_ + 15) / 16, 256, 0, stream>>>(dec_state, B_ * U1_, wdecp, b_dec, dec_h);

  // 3) fused joint + logits + log-softmax select (the 70 GFLOP WMMA kernel)
  joint_softmax_kernel<<<B_ * (T_ / 2) * UT, 512, 0, stream>>>(enc_h, dec_h, wprojp, targets, blankL, emitL);

  // 4) alpha DP (one wave per batch, barrier-free) + final loss
  rnnt_dp_kernel<<<B_, 32, 0, stream>>>(blankL, emitL, inputs_length, targets_length, finals);
  finalize_kernel<<<1, 1, 0, stream>>>(finals, (float*)d_out);
}
